// MAGNET_59090160058449
// MI455X (gfx1250) — matrix operand order repaired and checked by reading, not compiled, not run on hardware
//
#include <hip/hip_runtime.h>
#include <hip/hip_bf16.h>
#include <math.h>

// ---------------- model constants ----------------
#define MB      128     // batch
#define FEAT    100
#define EDIM    256
#define NHEAD   8
#define HDIM    32
#define NL      4
#define FFDIM   1024
#define SEQL    256
#define NNODES  10000
#define NEDGES  160000
#define NODE_D  128
#define EDGE_D  16
#define NCLS    2

typedef __attribute__((ext_vector_type(16))) _Float16 v16h;
typedef __attribute__((ext_vector_type(8)))  _Float16 v8h;
typedef __attribute__((ext_vector_type(4)))  _Float16 v4h;
typedef __attribute__((ext_vector_type(8)))  float    v8f;

// =====================================================================
// Tiled GEMM: C[M,N] = A[M,K] @ W[K,N] + bias, optional ReLU.
// fp32 in/out, f16 WMMA (v_wmma_f32_16x16x32_f16) with fp32 accumulate.
// Block tile 64x128, K-step 32, 256 threads = 8 waves, 4 WMMA tiles/wave
// (one A fragment reused across 4 B fragments).
// LDS rows padded to 40 halves (80 B): keeps 16-B alignment for
// ds_load_b128 fragment reads and makes 16-lane reads bank-conflict-free.
// =====================================================================
#define BM 64
#define BN 128
#define BK 32
#define LPAD (BK + 8)     // 40 halves = 80 bytes per row

__global__ __launch_bounds__(256)
void gemm_wmma_f16(const float* __restrict__ A, const float* __restrict__ W,
                   const float* __restrict__ bias, float* __restrict__ C,
                   int M, int N, int K, int relu) {
  __shared__ alignas(16) _Float16 As[BM][LPAD];    // 5.0 KB
  __shared__ alignas(16) _Float16 BsT[BN][LPAD];   // 10.0 KB (B stored transposed)

  const int tid  = threadIdx.x;
  const int wave = tid >> 5;
  const int lane = tid & 31;
  const int hi   = lane >> 4;          // wave32 lane half
  const int l16  = lane & 15;

  const int m0 = blockIdx.y * BM;
  const int n0 = blockIdx.x * BN;
  const int tm  = wave >> 1;           // 0..3  (M tile)
  const int tnb = (wave & 1) * 4;      // 0 or 4 (first of 4 N tiles)

  const bool fullM = (m0 + BM) <= M;
  const bool fullN = (n0 + BN) <= N;
  const bool alK4  = (K & 3) == 0;

  v8f acc[4] = {};

  // fast-path staging coordinates
  const int ar  = tid >> 2;            // A row 0..63
  const int ac  = (tid & 3) * 8;       // A col 0/8/16/24
  const int bc  = tid & 127;           // B column (N) 0..127
  const int bk0 = (tid >> 7) * 16;     // B K base 0 or 16

  for (int k0 = 0; k0 < K; k0 += BK) {
    const bool fast = fullM && fullN && alK4 && ((k0 + BK) <= K);
    if (fast) {
      // ---- A: 2x float4 per thread, packed to 2x ds_store_b64 ----
      const float* aptr = A + (size_t)(m0 + ar) * K + (k0 + ac);
      float4 av0 = *(const float4*)(aptr);
      float4 av1 = *(const float4*)(aptr + 4);
      v4h ah0 = { (_Float16)av0.x, (_Float16)av0.y, (_Float16)av0.z, (_Float16)av0.w };
      v4h ah1 = { (_Float16)av1.x, (_Float16)av1.y, (_Float16)av1.z, (_Float16)av1.w };
      *(v4h*)&As[ar][ac]     = ah0;
      *(v4h*)&As[ar][ac + 4] = ah1;
      if (k0 + BK < K) __builtin_prefetch(aptr + BK, 0, 1);
      // ---- B: coalesced b32 loads along N, transposed b64 stores ----
#pragma unroll
      for (int g = 0; g < 4; ++g) {
        int kk = bk0 + g * 4;
        const float* wp = W + (size_t)(k0 + kk) * N + (n0 + bc);
        float w0 = wp[0];
        float w1 = wp[(size_t)N];
        float w2 = wp[(size_t)2 * N];
        float w3 = wp[(size_t)3 * N];
        v4h bh = { (_Float16)w0, (_Float16)w1, (_Float16)w2, (_Float16)w3 };
        *(v4h*)&BsT[bc][kk] = bh;
      }
      if (k0 + BK < K)
        __builtin_prefetch(W + (size_t)(k0 + BK + bk0) * N + (n0 + bc), 0, 1);
    } else {
      // ---- guarded scalar path for edge tiles ----
      for (int i = tid; i < BM * BK; i += 256) {
        int r = i >> 5, c = i & 31;
        int gr = m0 + r, gc = k0 + c;
        float v = (gr < M && gc < K) ? A[(size_t)gr * K + gc] : 0.0f;
        As[r][c] = (_Float16)v;
      }
      for (int i = tid; i < BK * BN; i += 256) {
        int r = i >> 7, c = i & 127;   // r = K index, c = N index
        int gr = k0 + r, gc = n0 + c;
        float v = (gr < K && gc < N) ? W[(size_t)gr * N + gc] : 0.0f;
        BsT[c][r] = (_Float16)v;
      }
    }
    __syncthreads();

    // ---- fragments: two ds_load_b128 each (ISA 7.12.2 layouts) ----
    const _Float16* arow = &As[tm * 16 + l16][0];
    v8h a0 = *(const v8h*)(arow + hi * 8);        // K hi*8 .. hi*8+7
    v8h a1 = *(const v8h*)(arow + 16 + hi * 8);   // K 16+hi*8 ..
    v16h af = __builtin_shufflevector(a0, a1, 0, 1, 2, 3, 4, 5, 6, 7,
                                      8, 9, 10, 11, 12, 13, 14, 15);
#pragma unroll
    for (int j = 0; j < 4; ++j) {
      const _Float16* brow = &BsT[(tnb + j) * 16 + l16][0];
      v8h b0 = *(const v8h*)(brow + hi * 8);      // K hi*8 .. hi*8+7
      v8h b1 = *(const v8h*)(brow + 16 + hi * 8); // K 16+hi*8 ..
      v16h bf = __builtin_shufflevector(b0, b1, 0, 1, 2, 3, 4, 5, 6, 7,
                                        8, 9, 10, 11, 12, 13, 14, 15);
      acc[j] = __builtin_amdgcn_wmma_f32_16x16x32_f16(false, af, false, bf,
                                                      (short)0, acc[j], false, false);
    }
    __syncthreads();
  }

  // ---- epilogue: C layout VGPR r -> M=r (+8 for hi lanes), N = lane&15 ----
  const int row_base = m0 + tm * 16 + (hi << 3);
#pragma unroll
  for (int j = 0; j < 4; ++j) {
    int col = n0 + (tnb + j) * 16 + l16;
    if (col >= N) continue;
    float bb = bias ? bias[col] : 0.0f;
#pragma unroll
    for (int r = 0; r < 8; ++r) {
      int row = row_base + r;
      if (row >= M) continue;
      float v = acc[j][r] + bb;
      if (relu) v = v > 0.0f ? v : 0.0f;
      C[(size_t)row * N + col] = v;
    }
  }
}

// =====================================================================
// Fused attention: one workgroup per (b,h); K/V staged in LDS,
// online softmax per query row (thread). Q,K,V,O: [B,S,E], head = h*32.
// =====================================================================
__global__ __launch_bounds__(256)
void attn_fused(const float* __restrict__ Q, const float* __restrict__ K,
                const float* __restrict__ V, float* __restrict__ O,
                const float* __restrict__ imp, int S, float scale) {
  __shared__ float sK[SEQL * HDIM];   // 32 KB
  __shared__ float sV[SEQL * HDIM];   // 32 KB (320 KB LDS/WGP on CDNA5)
  const int bh = blockIdx.x;
  const int b = bh >> 3, h = bh & 7;
  const size_t base = ((size_t)b * S) * EDIM + h * HDIM;

  for (int i = threadIdx.x; i < S * HDIM; i += blockDim.x) {
    int s = i >> 5, d = i & 31;
    sK[i] = K[base + (size_t)s * EDIM + d];
    sV[i] = V[base + (size_t)s * EDIM + d];
  }
  __syncthreads();

  const int qi = threadIdx.x;
  if (qi >= S) return;

  float qr[HDIM];
#pragma unroll
  for (int d = 0; d < HDIM; ++d) qr[d] = Q[base + (size_t)qi * EDIM + d];
  const float im = imp[0];

  float m = -3.4e38f, l = 0.0f, o[HDIM];
#pragma unroll
  for (int d = 0; d < HDIM; ++d) o[d] = 0.0f;

  for (int j = 0; j < S; ++j) {
    float s = 0.0f;
#pragma unroll
    for (int d = 0; d < HDIM; ++d) s = fmaf(qr[d], sK[j * HDIM + d], s);
    s = s * scale * im;
    float mn = fmaxf(m, s);
    float corr = __expf(m - mn);
    float p = __expf(s - mn);
    l = l * corr + p;
#pragma unroll
    for (int d = 0; d < HDIM; ++d) o[d] = o[d] * corr + p * sV[j * HDIM + d];
    m = mn;
  }
  float inv = 1.0f / l;
#pragma unroll
  for (int d = 0; d < HDIM; ++d) O[base + (size_t)qi * EDIM + d] = o[d] * inv;
}

// ---------------- fusion (S=3) attention, also emits probs [B,H,3,3] ---
__global__ void fusion_attn_kernel(const float* __restrict__ Q, const float* __restrict__ K,
                                   const float* __restrict__ V, float* __restrict__ O,
                                   float* __restrict__ probs, const float* __restrict__ imp) {
  int t = blockIdx.x * blockDim.x + threadIdx.x;
  if (t >= MB * NHEAD * 3) return;
  int q = t % 3, h = (t / 3) & 7, b = t / 24;
  size_t base = ((size_t)b * 3) * EDIM + h * HDIM;
  const float im = imp[0];
  float s[3];
  for (int j = 0; j < 3; ++j) {
    float acc = 0.0f;
    for (int d = 0; d < HDIM; ++d)
      acc = fmaf(Q[base + (size_t)q * EDIM + d], K[base + (size_t)j * EDIM + d], acc);
    s[j] = acc * 0.17677669529663687f * im;   // 1/sqrt(32) * imp
  }
  float mx = fmaxf(s[0], fmaxf(s[1], s[2]));
  float e0 = __expf(s[0] - mx), e1 = __expf(s[1] - mx), e2 = __expf(s[2] - mx);
  float den = e0 + e1 + e2;
  float p[3] = { e0 / den, e1 / den, e2 / den };
  for (int j = 0; j < 3; ++j)
    probs[(((size_t)b * NHEAD + h) * 3 + q) * 3 + j] = p[j];
  for (int d = 0; d < HDIM; ++d) {
    float o = p[0] * V[base + d] + p[1] * V[base + EDIM + d] + p[2] * V[base + 2 * EDIM + d];
    O[base + (size_t)q * EDIM + d] = o;
  }
}

// ---------------- LayerNorm (E=256), wave-per-row, optional residual ----
__global__ __launch_bounds__(256)
void ln_kernel(float* __restrict__ out, const float* __restrict__ x,
               const float* __restrict__ res, const float* __restrict__ g,
               const float* __restrict__ b, int rows) {
  int row = blockIdx.x * 8 + (threadIdx.x >> 5);
  int lane = threadIdx.x & 31;
  if (row >= rows) return;
  const float* xr = x + (size_t)row * EDIM;
  const float* rr = res ? res + (size_t)row * EDIM : nullptr;
  float v[8], sum = 0.0f;
#pragma unroll
  for (int i = 0; i < 8; ++i) {
    int c = lane + i * 32;
    float t = xr[c] + (rr ? rr[c] : 0.0f);
    v[i] = t; sum += t;
  }
#pragma unroll
  for (int off = 16; off > 0; off >>= 1) sum += __shfl_xor(sum, off, 32);
  float mean = sum * (1.0f / 256.0f);
  float var = 0.0f;
#pragma unroll
  for (int i = 0; i < 8; ++i) { float d = v[i] - mean; var += d * d; }
#pragma unroll
  for (int off = 16; off > 0; off >>= 1) var += __shfl_xor(var, off, 32);
  float rstd = rsqrtf(var * (1.0f / 256.0f) + 1e-5f);
  float* orow = out + (size_t)row * EDIM;
#pragma unroll
  for (int i = 0; i < 8; ++i) {
    int c = lane + i * 32;
    orow[c] = (v[i] - mean) * rstd * g[c] + b[c];
  }
}

// ---------------- small elementwise / embedding kernels -----------------
__global__ void tab_embed_kernel(const float* __restrict__ tx, const float* __restrict__ w,
                                 const float* __restrict__ b, float* __restrict__ out, int rows) {
  int t = blockIdx.x * blockDim.x + threadIdx.x;
  if (t >= rows * EDIM) return;
  int row = t >> 8, e = t & 255;
  out[t] = tx[row] * w[e] + b[e];
}

__global__ void relu_add_pos_kernel(float* __restrict__ x, const float* __restrict__ pos, int n) {
  int t = blockIdx.x * blockDim.x + threadIdx.x;
  if (t >= n) return;
  int e = t & 255, f = (t >> 8) % FEAT;
  float v = x[t];
  x[t] = (v > 0.0f ? v : 0.0f) + pos[f * EDIM + e];
}

__global__ void seq_embed_kernel(const int* __restrict__ seq, const float* __restrict__ table,
                                 float* __restrict__ out, int n) {
  int t = blockIdx.x * blockDim.x + threadIdx.x;
  if (t >= n) return;
  int e = t & 255, bl = t >> 8, l = bl % SEQL;
  int tok = seq[bl];
  float emb = (tok == 0) ? 0.0f : table[(size_t)tok * EDIM + e];
  int i2 = e >> 1;
  float freq = __expf(-(float)(2 * i2) * (logf(10000.0f) / (float)EDIM));
  float ang = (float)l * freq;
  float pe = (e & 1) ? __cosf(ang) : __sinf(ang);
  out[t] = emb + pe;
}

__global__ void mean_axis1_kernel(const float* __restrict__ x, float* __restrict__ out,
                                  int B, int S) {
  int t = blockIdx.x * blockDim.x + threadIdx.x;
  if (t >= B * EDIM) return;
  int b = t >> 8, e = t & 255;
  float s = 0.0f;
  for (int i = 0; i < S; ++i) s += x[((size_t)b * S + i) * EDIM + e];
  out[t] = s / (float)S;
}

__global__ void mean_axis0_kernel(const float* __restrict__ x, float* __restrict__ out, int N) {
  int e = blockIdx.x * blockDim.x + threadIdx.x;
  if (e >= EDIM) return;
  float s = 0.0f;
  for (int i = 0; i < N; ++i) s += x[(size_t)i * EDIM + e];
  out[e] = s / (float)N;
}

__global__ void fill_kernel(float* __restrict__ p, float v, size_t n) {
  size_t t = (size_t)blockIdx.x * blockDim.x + threadIdx.x;
  if (t < n) p[t] = v;
}

// ---------------- graph edge-softmax kernels ---------------------------
__device__ __forceinline__ void atomicMaxF(float* addr, float val) {
  int old = __float_as_int(*addr);
  while (__int_as_float(old) < val) {
    int assumed = old;
    old = atomicCAS((int*)addr, assumed, __float_as_int(val));
    if (old == assumed) break;
  }
}

__global__ void edge_score_kernel(const float* __restrict__ q, const float* __restrict__ k,
                                  const float* __restrict__ ep, const int* __restrict__ src,
                                  const int* __restrict__ dst, float* __restrict__ score, int Ne) {
  int t = blockIdx.x * blockDim.x + threadIdx.x;
  if (t >= Ne * NHEAD) return;
  int e = t >> 3, h = t & 7;
  int s = src[e], d = dst[e];
  const float* qr = q + (size_t)d * EDIM + h * HDIM;
  const float* kr = k + (size_t)s * EDIM + h * HDIM;
  const float* er = ep + (size_t)e * EDIM + h * HDIM;
  float acc = 0.0f;
#pragma unroll
  for (int i = 0; i < HDIM; ++i) acc = fmaf(qr[i], kr[i] + er[i], acc);
  score[t] = acc * 0.17677669529663687f;   // 1/sqrt(32)
}

__global__ void seg_max_kernel(const float* __restrict__ score, const int* __restrict__ dst,
                               float* __restrict__ smax, int Ne) {
  int t = blockIdx.x * blockDim.x + threadIdx.x;
  if (t >= Ne * NHEAD) return;
  int e = t >> 3, h = t & 7;
  atomicMaxF(&smax[dst[e] * NHEAD + h], score[t]);
}

__global__ void seg_expsum_kernel(float* __restrict__ score, const int* __restrict__ dst,
                                  const float* __restrict__ smax, float* __restrict__ den, int Ne) {
  int t = blockIdx.x * blockDim.x + threadIdx.x;
  if (t >= Ne * NHEAD) return;
  int e = t >> 3, h = t & 7;
  float ex = __expf(score[t] - smax[dst[e] * NHEAD + h]);
  score[t] = ex;
  atomicAdd(&den[dst[e] * NHEAD + h], ex);
}

__global__ void seg_agg_kernel(const float* __restrict__ ex, const float* __restrict__ den,
                               const float* __restrict__ v, const float* __restrict__ ep,
                               const int* __restrict__ src, const int* __restrict__ dst,
                               float* __restrict__ agg, int Ne) {
  int t = blockIdx.x * blockDim.x + threadIdx.x;
  if (t >= Ne * NHEAD) return;
  int e = t >> 3, h = t & 7;
  int s = src[e], d = dst[e];
  float alpha = ex[t] / (den[d * NHEAD + h] + 1e-16f);
  const float* vr = v + (size_t)s * EDIM + h * HDIM;
  const float* er = ep + (size_t)e * EDIM + h * HDIM;
  float* ar = agg + (size_t)d * EDIM + h * HDIM;
#pragma unroll
  for (int i = 0; i < HDIM; ++i) atomicAdd(&ar[i], alpha * (vr[i] + er[i]));
}

__global__ void node_update_kernel(float* __restrict__ xg, const float* __restrict__ agg,
                                   const float* __restrict__ skip, size_t n) {
  size_t t = (size_t)blockIdx.x * blockDim.x + threadIdx.x;
  if (t >= n) return;
  float a = agg[t] + skip[t];
  xg[t] += (a > 0.0f ? a : 0.0f);
}

// ---------------- fusion helpers ---------------------------------------
__global__ void stack3_kernel(const float* __restrict__ te, const float* __restrict__ ge,
                              const float* __restrict__ se, float* __restrict__ comb) {
  int t = blockIdx.x * blockDim.x + threadIdx.x;
  if (t >= MB * 3 * EDIM) return;
  int e = t & 255, s = (t >> 8) % 3, b = t / (3 * EDIM);
  float v = (s == 0) ? te[b * EDIM + e] : (s == 1) ? ge[e] : se[b * EDIM + e];
  comb[t] = v;
}

__global__ void mean3_kernel(const float* __restrict__ c, float* __restrict__ out) {
  int t = blockIdx.x * blockDim.x + threadIdx.x;
  if (t >= MB * EDIM) return;
  int b = t >> 8, e = t & 255;
  size_t base = (size_t)b * 3 * EDIM + e;
  out[t] = (c[base] + c[base + EDIM] + c[base + 2 * EDIM]) * (1.0f / 3.0f);
}

// =====================================================================
// Host-side orchestration
// =====================================================================
struct LinP  { const float* w; const float* b; };
struct LnP   { const float* g; const float* b; };
struct AttnP { LinP q, k, v, o; const float* imp; };
struct BlockP{ AttnP attn; LnP ln1; LinP ff1, ff2; LnP ln2; };
struct ConvP { LinP q, k, v, e, skip; };

static inline int cdiv(int a, int b) { return (a + b - 1) / b; }

static void gemm(const float* A, const LinP& w, float* C, int M, int N, int K,
                 int relu, hipStream_t st) {
  dim3 grid(cdiv(N, BN), cdiv(M, BM));
  gemm_wmma_f16<<<grid, 256, 0, st>>>(A, w.w, w.b, C, M, N, K, relu);
}

// one transformer encoder block stack over x [rows=B*S, 256]
static void run_transformer(float* x, int B, int S, const BlockP* blk, int nb,
                            float* bq, float* bk, float* bv, float* bab, float* bh,
                            hipStream_t st) {
  const int rows = B * S;
  const float scale = 0.17677669529663687f;   // 1/sqrt(HD)
  for (int i = 0; i < nb; ++i) {
    const BlockP& p = blk[i];
    gemm(x, p.attn.q, bq, rows, EDIM, EDIM, 0, st);
    gemm(x, p.attn.k, bk, rows, EDIM, EDIM, 0, st);
    gemm(x, p.attn.v, bv, rows, EDIM, EDIM, 0, st);
    attn_fused<<<B * NHEAD, 256, 0, st>>>(bq, bk, bv, bab, p.attn.imp, S, scale);
    gemm(bab, p.attn.o, bq, rows, EDIM, EDIM, 0, st);                 // bq = attn out proj
    ln_kernel<<<cdiv(rows, 8), 256, 0, st>>>(bk, x, bq, p.ln1.g, p.ln1.b, rows);  // bk = x1
    gemm(bk, p.ff1, bh, rows, FFDIM, EDIM, 1, st);                    // relu fused
    gemm(bh, p.ff2, bv, rows, EDIM, FFDIM, 0, st);                    // bv = ff out
    ln_kernel<<<cdiv(rows, 8), 256, 0, st>>>(x, bk, bv, p.ln2.g, p.ln2.b, rows);
  }
}

extern "C" void kernel_launch(void* const* d_in, const int* in_sizes, int n_in,
                              void* d_out, int out_size, void* d_ws, size_t ws_size,
                              hipStream_t stream) {
  (void)in_sizes; (void)n_in; (void)out_size; (void)ws_size;
  int pi = 0;
  const float* tab_x      = (const float*)d_in[pi++];
  const float* graph_x    = (const float*)d_in[pi++];
  const int*   edge_index = (const int*)  d_in[pi++];
  const float* edge_attr  = (const float*)d_in[pi++];
  const int*   seq        = (const int*)  d_in[pi++];

  auto F       = [&]() { return (const float*)d_in[pi++]; };
  auto getLin  = [&]() { LinP l; l.w = F(); l.b = F(); return l; };
  auto getLn   = [&]() { LnP l; l.g = F(); l.b = F(); return l; };
  auto getAttn = [&]() { AttnP a; a.q=getLin(); a.k=getLin(); a.v=getLin(); a.o=getLin(); a.imp=F(); return a; };
  auto getBlk  = [&]() { BlockP b; b.attn=getAttn(); b.ln1=getLn(); b.ff1=getLin(); b.ff2=getLin(); b.ln2=getLn(); return b; };
  auto getConv = [&]() { ConvP c; c.q=getLin(); c.k=getLin(); c.v=getLin(); c.e=getLin(); c.skip=getLin(); return c; };

  // --- params (setup_inputs insertion order) ---
  LinP tab_emb_l = getLin(); LnP tab_emb_ln = getLn(); const float* tab_pos = F();
  BlockP tab_blocks[NL];  for (int i = 0; i < NL; ++i) tab_blocks[i] = getBlk();
  LinP node_emb = getLin(); LinP edge_emb = getLin();
  ConvP convs[NL];        for (int i = 0; i < NL; ++i) convs[i] = getConv();
  LnP gnorm = getLn(); LinP out_proj = getLin();
  const float* table = F();
  BlockP seq_blocks[NL];  for (int i = 0; i < NL; ++i) seq_blocks[i] = getBlk();
  AttnP f_attn = getAttn(); LnP f_ln1 = getLn();
  LinP f_ff1 = getLin(); LinP f_ff2 = getLin(); LnP f_ln2 = getLn();
  LinP f_ssl = getLin(); LinP f_cls = getLin();

  // --- workspace bump allocator ---
  char* ws = (char*)d_ws;
  size_t off = 0;
  auto alloc = [&](size_t bytes) -> float* {
    bytes = (bytes + 255) & ~(size_t)255;
    float* p = (float*)(ws + off);
    off += bytes;
    return p;
  };

  // persistent region
  float* tab_emb  = alloc((size_t)MB * EDIM * 4);
  float* graph_emb= alloc((size_t)EDIM * 4);
  float* seq_emb  = alloc((size_t)MB * EDIM * 4);
  float* comb     = alloc((size_t)MB * 3 * EDIM * 4);
  float* f_q      = alloc((size_t)MB * 3 * EDIM * 4);
  float* f_k      = alloc((size_t)MB * 3 * EDIM * 4);
  float* f_v      = alloc((size_t)MB * 3 * EDIM * 4);
  float* f_att    = alloc((size_t)MB * 3 * EDIM * 4);
  float* f_h      = alloc((size_t)MB * 3 * FFDIM * 4);
  float* fused    = alloc((size_t)MB * EDIM * 4);
  float* gmean    = alloc((size_t)EDIM * 4);
  const size_t arena = off;

  // ==================== tabular branch ====================
  {
    off = arena;
    const int rows = MB * FEAT;                 // 12800
    float* x  = alloc((size_t)rows * EDIM * 4);
    float* bq = alloc((size_t)rows * EDIM * 4);
    float* bk = alloc((size_t)rows * EDIM * 4);
    float* bv = alloc((size_t)rows * EDIM * 4);
    float* ba = alloc((size_t)rows * EDIM * 4);
    float* bh = alloc((size_t)rows * FFDIM * 4);
    tab_embed_kernel<<<cdiv(rows * EDIM, 256), 256, 0, stream>>>(tab_x, tab_emb_l.w, tab_emb_l.b, x, rows);
    ln_kernel<<<cdiv(rows, 8), 256, 0, stream>>>(x, x, nullptr, tab_emb_ln.g, tab_emb_ln.b, rows);
    relu_add_pos_kernel<<<cdiv(rows * EDIM, 256), 256, 0, stream>>>(x, tab_pos, rows * EDIM);
    run_transformer(x, MB, FEAT, tab_blocks, NL, bq, bk, bv, ba, bh, stream);
    mean_axis1_kernel<<<cdiv(MB * EDIM, 256), 256, 0, stream>>>(x, tab_emb, MB, FEAT);
  }

  // ==================== graph branch ====================
  {
    off = arena;
    float* xg    = alloc((size_t)NNODES * EDIM * 4);
    float* ea    = alloc((size_t)NEDGES * EDIM * 4);   // edge embedding (persistent in branch)
    float* ep    = alloc((size_t)NEDGES * EDIM * 4);   // per-layer e-projection
    float* gq    = alloc((size_t)NNODES * EDIM * 4);
    float* gk    = alloc((size_t)NNODES * EDIM * 4);
    float* gv    = alloc((size_t)NNODES * EDIM * 4);
    float* gskip = alloc((size_t)NNODES * EDIM * 4);
    float* gagg  = alloc((size_t)NNODES * EDIM * 4);
    float* score = alloc((size_t)NEDGES * NHEAD * 4);
    float* smax  = alloc((size_t)NNODES * NHEAD * 4);
    float* den   = alloc((size_t)NNODES * NHEAD * 4);
    const int* src = edge_index;
    const int* dst = edge_index + NEDGES;

    gemm(graph_x,   node_emb, xg, NNODES, EDIM, NODE_D, 0, stream);
    gemm(edge_attr, edge_emb, ea, NEDGES, EDIM, EDGE_D, 0, stream);

    for (int l = 0; l < NL; ++l) {
      const ConvP& c = convs[l];
      gemm(xg, c.q,    gq,    NNODES, EDIM, EDIM, 0, stream);
      gemm(xg, c.k,    gk,    NNODES, EDIM, EDIM, 0, stream);
      gemm(xg, c.v,    gv,    NNODES, EDIM, EDIM, 0, stream);
      gemm(xg, c.skip, gskip, NNODES, EDIM, EDIM, 0, stream);
      gemm(ea, c.e,    ep,    NEDGES, EDIM, EDIM, 0, stream);
      fill_kernel<<<cdiv(NNODES * NHEAD, 256), 256, 0, stream>>>(smax, -3.4e38f, (size_t)NNODES * NHEAD);
      fill_kernel<<<cdiv(NNODES * NHEAD, 256), 256, 0, stream>>>(den, 0.0f, (size_t)NNODES * NHEAD);
      fill_kernel<<<cdiv(NNODES * EDIM, 256), 256, 0, stream>>>(gagg, 0.0f, (size_t)NNODES * EDIM);
      const int nt = NEDGES * NHEAD;
      edge_score_kernel<<<cdiv(nt, 256), 256, 0, stream>>>(gq, gk, ep, src, dst, score, NEDGES);
      seg_max_kernel<<<cdiv(nt, 256), 256, 0, stream>>>(score, dst, smax, NEDGES);
      seg_expsum_kernel<<<cdiv(nt, 256), 256, 0, stream>>>(score, dst, smax, den, NEDGES);
      seg_agg_kernel<<<cdiv(nt, 256), 256, 0, stream>>>(score, den, gv, ep, src, dst, gagg, NEDGES);
      node_update_kernel<<<cdiv(NNODES * EDIM, 256), 256, 0, stream>>>(xg, gagg, gskip, (size_t)NNODES * EDIM);
    }
    ln_kernel<<<cdiv(NNODES, 8), 256, 0, stream>>>(gk, xg, nullptr, gnorm.g, gnorm.b, NNODES);
    mean_axis0_kernel<<<1, 256, 0, stream>>>(gk, gmean, NNODES);
    gemm(gmean, out_proj, graph_emb, 1, EDIM, EDIM, 0, stream);
  }

  // ==================== sequence branch ====================
  {
    off = arena;
    const int rows = MB * SEQL;                 // 32768
    float* x  = alloc((size_t)rows * EDIM * 4);
    float* bq = alloc((size_t)rows * EDIM * 4);
    float* bk = alloc((size_t)rows * EDIM * 4);
    float* bv = alloc((size_t)rows * EDIM * 4);
    float* ba = alloc((size_t)rows * EDIM * 4);
    float* bh = alloc((size_t)rows * FFDIM * 4);
    seq_embed_kernel<<<cdiv(rows * EDIM, 256), 256, 0, stream>>>(seq, table, x, rows * EDIM);
    run_transformer(x, MB, SEQL, seq_blocks, NL, bq, bk, bv, ba, bh, stream);
    mean_axis1_kernel<<<cdiv(MB * EDIM, 256), 256, 0, stream>>>(x, seq_emb, MB, SEQL);
  }

  // ==================== fusion ====================
  {
    float* out_cls  = (float*)d_out;                       // [128, 2]
    float* out_ssl  = (float*)d_out + MB * NCLS;           // [128, 256]
    float* out_attn = out_ssl + (size_t)MB * EDIM;         // [128, 8, 3, 3]
    const int rows = MB * 3;

    stack3_kernel<<<cdiv(rows * EDIM, 256), 256, 0, stream>>>(tab_emb, graph_emb, seq_emb, comb);
    gemm(comb, f_attn.q, f_q, rows, EDIM, EDIM, 0, stream);
    gemm(comb, f_attn.k, f_k, rows, EDIM, EDIM, 0, stream);
    gemm(comb, f_attn.v, f_v, rows, EDIM, EDIM, 0, stream);
    fusion_attn_kernel<<<cdiv(rows * NHEAD, 256), 256, 0, stream>>>(f_q, f_k, f_v, f_att, out_attn, f_attn.imp);
    gemm(f_att, f_attn.o, f_q, rows, EDIM, EDIM, 0, stream);                       // f_q = o-proj
    ln_kernel<<<cdiv(rows, 8), 256, 0, stream>>>(f_k, comb, f_q, f_ln1.g, f_ln1.b, rows);  // f_k = c
    gemm(f_k, f_ff1, f_h, rows, 4 * EDIM, EDIM, 1, stream);
    gemm(f_h, f_ff2, f_v, rows, EDIM, 4 * EDIM, 0, stream);
    ln_kernel<<<cdiv(rows, 8), 256, 0, stream>>>(comb, f_k, f_v, f_ln2.g, f_ln2.b, rows);  // comb = c2
    mean3_kernel<<<cdiv(MB * EDIM, 256), 256, 0, stream>>>(comb, fused);
    gemm(fused, f_cls, out_cls, MB, NCLS, EDIM, 0, stream);
    gemm(fused, f_ssl, out_ssl, MB, EDIM, EDIM, 0, stream);
  }
}
